// RNN_19576460935309
// MI455X (gfx1250) — matrix operand order repaired and checked by reading, not compiled, not run on hardware
//
#include <hip/hip_runtime.h>

// LSTM: T=8192 sequential steps, B=128, H=96, IN=1.
// Persistent single-workgroup kernel on one WGP, sized to the VGPR file:
//  - 12 waves (384 thr) = 3 waves/SIMD -> 256-VGPR budget: W_hh f16 B-frags
//    (96 VGPRs) stay resident with zero spills
//  - wave = (mh 0..1, k 0..5): owns M-tiles {4mh..4mh+3} and the four gate
//    N-tiles {k,6+k,12+k,18+k}; 4 passes/step reuse one 32-VGPR accumulator set
//  - h double-buffered in LDS (ping-pong by t&1) -> one barrier/step,
//    A-fragments loaded lazily (8 live VGPRs)
//  - v_tanh_f32 for all gate nonlinearities (1 TRANS per gate)
//  - fused x-projection; fused output projection via LDS f32 atomics
//  - x staged one step ahead in LDS, global x prefetched two steps ahead

typedef _Float16 half16 __attribute__((ext_vector_type(16)));
typedef _Float16 half8  __attribute__((ext_vector_type(8)));
typedef float    float8 __attribute__((ext_vector_type(8)));

union AFrag { half16 v; half8 h[2]; };

#if defined(__has_builtin)
#if __has_builtin(__builtin_amdgcn_tanhf)
#define HAVE_HW_TANH 1
#endif
#endif

__device__ __forceinline__ float tanh_fast(float z) {
#ifdef HAVE_HW_TANH
    return __builtin_amdgcn_tanhf(z);                       // v_tanh_f32
#else
    return __builtin_amdgcn_rcpf(1.0f + __builtin_amdgcn_exp2f(-2.88539008f * z)) * 2.0f - 1.0f;
#endif
}
__device__ __forceinline__ float sigm_fast(float z) {
#ifdef HAVE_HW_TANH
    return __builtin_fmaf(0.5f, __builtin_amdgcn_tanhf(0.5f * z), 0.5f);
#else
    return __builtin_amdgcn_rcpf(1.0f + __builtin_amdgcn_exp2f(-1.44269504f * z));
#endif
}

#define T_STEPS 8192
#define BATCH   128
#define HID     96
#define HPAD    104   // row stride in halves: 208 B = 16*13 -> b128-aligned rows
#define NTHREADS 384

__global__ __launch_bounds__(NTHREADS, 1)
void lstm_persistent(const float* __restrict__ x,      // (T,B,1)
                     const float* __restrict__ W_ih,   // (384,1)
                     const float* __restrict__ W_hh,   // (384,96)
                     const float* __restrict__ b_ih,   // (384)
                     const float* __restrict__ b_hh,   // (384)
                     const float* __restrict__ W_fc,   // (1,96)
                     const float* __restrict__ b_fc,   // (1)
                     float* __restrict__ out)          // (T,B,1)
{
    __shared__ __align__(16) _Float16 h_lds[2][BATCH][HPAD];
    __shared__ __align__(16) float    x_lds[2][BATCH];
    __shared__ __align__(16) float    outAcc[2][BATCH];

    const int tid  = threadIdx.x;
    const int wave = tid >> 5;        // 0..11
    const int lane = tid & 31;
    const int k    = wave % 6;        // h-column block (16 cols)
    const int mh   = wave / 6;        // half of the M (batch) tiles: 0..1
    const int ncol = lane & 15;
    const int hi   = lane >> 4;       // half-wave select

    // ---- per-lane constants ----
    float wih[4], bias[4];
#pragma unroll
    for (int G = 0; G < 4; ++G) {
        int g = G * 96 + k * 16 + ncol;
        wih[G]  = W_ih[g];
        bias[G] = b_ih[g] + b_hh[g];
    }
    const float wfc = W_fc[k * 16 + ncol];
    const float bfc = b_fc[0];

    // ---- resident W_hh^T B-fragments (f16), one-time load ----
    // B 32x16 layout: lane%16 = column (gate row), elements 0-7 hold K=8*hi+e,
    // elements 8-15 hold K=16+8*hi+(e-8), same order as the A-frag.
    half16 bf[4][3];
#pragma unroll
    for (int G = 0; G < 4; ++G) {
#pragma unroll
        for (int kt = 0; kt < 3; ++kt) {
            const float* wp = W_hh + (G * 96 + k * 16 + ncol) * HID + kt * 32 + hi * 8;
            half16 b;
#pragma unroll
            for (int e = 0; e < 16; ++e) b[e] = (_Float16)wp[e < 8 ? e : e + 8];
            bf[G][kt] = b;
        }
    }

    // ---- init shared state ----
    for (int i = tid; i < 2 * BATCH * HPAD; i += NTHREADS) ((_Float16*)h_lds)[i] = (_Float16)0.0f;
    if (tid < 2 * BATCH) ((float*)outAcc)[tid] = 0.0f;

    float c[4][8];
#pragma unroll
    for (int mi = 0; mi < 4; ++mi)
#pragma unroll
        for (int r = 0; r < 8; ++r) c[mi][r] = 0.0f;

    float4 xf4;
    if (wave == 0) {
        float4 x0 = *(const float4*)(x + lane * 4);              // x[0]
        *(float4*)&x_lds[0][lane * 4] = x0;
        xf4 = *(const float4*)(x + BATCH + lane * 4);            // x[1]
    }
    __syncthreads();

    for (int t = 0; t < T_STEPS; ++t) {
        const int cur = t & 1;
        const _Float16* hs = &h_lds[cur][0][0];        // read h[t]
        _Float16*       hd = &h_lds[cur ^ 1][0][0];    // write h[t+1]
        const float*    xs = &x_lds[cur][0];           // x[t] (staged last step)
        float*          oA = &outAcc[cur][0];          // this step's accumulator

        if (wave == 0) {
            if (t > 0) {   // drain step t-1: outAcc[(t-1)&1] + bias + x[t-1]
                const int pb = (t - 1) & 1;
                float4 xp = *(const float4*)&x_lds[pb][lane * 4];
                float4 oa = *(const float4*)&outAcc[pb][lane * 4];
                float4 o4;
                o4.x = oa.x + bfc + xp.x;  o4.y = oa.y + bfc + xp.y;
                o4.z = oa.z + bfc + xp.z;  o4.w = oa.w + bfc + xp.w;
                *(float4*)(out + (size_t)(t - 1) * BATCH + lane * 4) = o4;
                float4 z4 = make_float4(0.f, 0.f, 0.f, 0.f);
                *(float4*)&outAcc[pb][lane * 4] = z4;  // reused at step t+1
            }
            if (t + 1 < T_STEPS)   // stage x[t+1] for next step
                *(float4*)&x_lds[(t + 1) & 1][lane * 4] = xf4;
            if (t + 2 < T_STEPS)   // prefetch x[t+2] from global
                xf4 = *(const float4*)(x + (size_t)(t + 2) * BATCH + lane * 4);
        }

        // ---- four passes over the wave's four M-tiles: one accumulator set ----
#pragma unroll
        for (int mi = 0; mi < 4; ++mi) {
            const int mb = (mh * 4 + mi) * 16;

            float8 acc[4];
#pragma unroll
            for (int G = 0; G < 4; ++G) { float8 z = {}; acc[G] = z; }

#pragma unroll
            for (int kt = 0; kt < 3; ++kt) {
                AFrag af;                                       // lazy A-frag: 8 VGPRs
                const int br  = mb + ncol;                      // A layout: M = lane%16
                const int off = kt * 32 + hi * 8;               // upper half: K+8 / K+24
                af.h[0] = *(const half8*)&hs[br * HPAD + off];
                af.h[1] = *(const half8*)&hs[br * HPAD + off + 16];
#pragma unroll
                for (int G = 0; G < 4; ++G)
                    acc[G] = __builtin_amdgcn_wmma_f32_16x16x32_f16(
                        false, af.v, false, bf[G][kt],
                        (short)0, acc[G], false, false);
            }

            // x values for this pass's 8 batch rows
            const int bb = mb + hi * 8;
            float4 a0 = *(const float4*)&xs[bb];
            float4 a1 = *(const float4*)&xs[bb + 4];
            float xr[8] = { a0.x, a0.y, a0.z, a0.w, a1.x, a1.y, a1.z, a1.w };

            // gate nonlinearities + c/h update (all 4 gates in-lane, element r)
#pragma unroll
            for (int r = 0; r < 8; ++r) {
                const int b = mb + hi * 8 + r;                  // C layout: M = r + 8*hi
                const float xv = xr[r];
                float zi = acc[0][r] + xv * wih[0] + bias[0];
                float zf = acc[1][r] + xv * wih[1] + bias[1];
                float zg = acc[2][r] + xv * wih[2] + bias[2];
                float zo = acc[3][r] + xv * wih[3] + bias[3];
                float ig = sigm_fast(zi), fg = sigm_fast(zf);
                float gg = tanh_fast(zg), og = sigm_fast(zo);
                float cn = fg * c[mi][r] + ig * gg;
                c[mi][r] = cn;
                float hn = og * tanh_fast(cn);
                hd[b * HPAD + k * 16 + ncol] = (_Float16)hn;    // next step's A operand
                atomicAdd(&oA[b], hn * wfc);                    // ds_add_f32
            }
        }
        __syncthreads();   // h[t+1] writes, out atomics, x staging all complete
    }

    // final timestep output
    if (wave == 0) {
        const int pb = (T_STEPS - 1) & 1;
        float4 xp = *(const float4*)&x_lds[pb][lane * 4];
        float4 oa = *(const float4*)&outAcc[pb][lane * 4];
        float4 o4;
        o4.x = oa.x + bfc + xp.x;  o4.y = oa.y + bfc + xp.y;
        o4.z = oa.z + bfc + xp.z;  o4.w = oa.w + bfc + xp.w;
        *(float4*)(out + (size_t)(T_STEPS - 1) * BATCH + lane * 4) = o4;
    }
}

extern "C" void kernel_launch(void* const* d_in, const int* in_sizes, int n_in,
                              void* d_out, int out_size, void* d_ws, size_t ws_size,
                              hipStream_t stream) {
    (void)in_sizes; (void)n_in; (void)out_size; (void)d_ws; (void)ws_size;
    const float* x    = (const float*)d_in[0];
    const float* W_ih = (const float*)d_in[1];
    const float* W_hh = (const float*)d_in[2];
    const float* b_ih = (const float*)d_in[3];
    const float* b_hh = (const float*)d_in[4];
    const float* W_fc = (const float*)d_in[5];
    const float* b_fc = (const float*)d_in[6];
    float* out = (float*)d_out;

    lstm_persistent<<<dim3(1), dim3(NTHREADS), 0, stream>>>(x, W_ih, W_hh, b_ih, b_hh, W_fc, b_fc, out);
}
